// MultiHeadSelfAttention_19043884990757
// MI455X (gfx1250) — matrix operand order repaired and checked by reading, not compiled, not run on hardware
//
#include <hip/hip_runtime.h>
#include <hip/hip_bf16.h>

typedef __attribute__((ext_vector_type(16))) _Float16 v16h;
typedef __attribute__((ext_vector_type(8)))  _Float16 v8h;
typedef __attribute__((ext_vector_type(8)))  float    v8f;
typedef __attribute__((ext_vector_type(4)))  unsigned int v4u;
typedef __attribute__((ext_vector_type(8)))  int      v8i;
typedef __attribute__((ext_vector_type(4)))  int      v4i;

#define D_MODEL 768
#define N_HEADS 12
#define D_K     64
#define BATCH   8
#define SEQ     1024
#define M_TOTAL (BATCH * SEQ)

#if __has_builtin(__builtin_amdgcn_tensor_load_to_lds) && \
    __has_builtin(__builtin_amdgcn_s_wait_tensorcnt)
#define USE_TDM 1
#else
#define USE_TDM 0
#endif

static __device__ __forceinline__ v16h hcat(v8h lo, v8h hi) {
  return __builtin_shufflevector(lo, hi, 0,1,2,3,4,5,6,7,8,9,10,11,12,13,14,15);
}

static __device__ __forceinline__ v8f wmma_f16(v16h a, v16h b, v8f c) {
  return __builtin_amdgcn_wmma_f32_16x16x32_f16(
      /*neg_a=*/false, a, /*neg_b=*/false, b,
      /*c_mod=*/(short)0, c, /*reuse_a=*/false, /*reuse_b=*/false);
}

// ---------------------------------------------------------------------------
// Stage one 32-key x 64-feature f16 tile (row stride D_MODEL elements) into
// LDS.  Preferred path: Tensor Data Mover (async, TENSORcnt); fallback: plain
// 128-bit copy loop.
// ---------------------------------------------------------------------------
static __device__ __forceinline__ void stage_tile(const _Float16* gsrc,
                                                  _Float16* ldst, int lane) {
#if USE_TDM
  // D# group 0: count=1 | lds_addr | 57-bit global addr | type=2 (bits 127:126)
  const unsigned lds_addr = (unsigned)(uintptr_t)ldst;   // generic addr[31:0] = LDS offset
  const unsigned long long ga = (unsigned long long)(uintptr_t)gsrc;
  const v4u g0 = { 1u, lds_addr, (unsigned)ga,
                   ((unsigned)(ga >> 32) & 0x01FFFFFFu) | 0x80000000u };
  // D# group 1: mask=0, data_size=1(2B) | tensor_dim0=64 | tensor_dim1=32 |
  //             tile_dim0=64 | tile_dim1=32, tile_dim2=0 | stride0=768
  const v8i g1 = { 0x10000,
                   (int)(64u << 16),   // [63:48]  tensor_dim0 lo
                   (int)(32u << 16),   // [95:80]  tensor_dim1 lo
                   (int)(64u << 16),   // [127:112] tile_dim0
                   32,                 // [143:128] tile_dim1
                   (int)D_MODEL,       // [191:160] tensor_dim0_stride lo
                   0, 0 };
  const v4i gz = { 0, 0, 0, 0 };
#if defined(__clang_major__) && (__clang_major__ >= 23)
  const v8i gz8 = { 0, 0, 0, 0, 0, 0, 0, 0 };
  __builtin_amdgcn_tensor_load_to_lds(g0, g1, gz, gz, gz8, 0);
#else
  __builtin_amdgcn_tensor_load_to_lds(g0, g1, gz, gz, 0);
#endif
  (void)lane;
#else
#pragma unroll
  for (int i = 0; i < 8; ++i) {
    const int c = i * 32 + lane;            // 256 x 16B chunks
    const int key = c >> 3, part = c & 7;
    ((uint4*)ldst)[c] = ((const uint4*)(gsrc + (size_t)key * D_MODEL))[part];
  }
#endif
}

static __device__ __forceinline__ void wait_tiles_pending_2() {
#if USE_TDM
  __builtin_amdgcn_s_wait_tensorcnt(2);
#endif
}
static __device__ __forceinline__ void wait_tiles_pending_0() {
#if USE_TDM
  __builtin_amdgcn_s_wait_tensorcnt(0);
#endif
}

// ---------------------------------------------------------------------------
// Kernel 1: fused QKV projection.  Y = X @ W^T + b, f16 out [M, 768].
// One wave per 16x64 output tile: A-frag reused across 4 N-subtiles.
// ---------------------------------------------------------------------------
__global__ __launch_bounds__(32) void qkv_proj_kernel(
    const float* __restrict__ X,
    const float* __restrict__ Wq, const float* __restrict__ bq,
    const float* __restrict__ Wk, const float* __restrict__ bk,
    const float* __restrict__ Wv, const float* __restrict__ bv,
    _Float16* __restrict__ qo, _Float16* __restrict__ ko,
    _Float16* __restrict__ vo)
{
  const int lane   = threadIdx.x;
  const int laneLo = lane & 15;
  const int laneHi = lane >> 4;
  const int nT = blockIdx.x;          // 0..11  (64-wide N tile)
  const int mT = blockIdx.y;          // 0..511 (16-wide M tile)
  const int which = blockIdx.z;       // 0=q 1=k 2=v

  const float* W    = (which == 0) ? Wq : ((which == 1) ? Wk : Wv);
  const float* bias = (which == 0) ? bq : ((which == 1) ? bk : bv);
  _Float16*    out  = (which == 0) ? qo : ((which == 1) ? ko : vo);

  const float* xrow = X + (size_t)(mT * 16 + laneLo) * D_MODEL;
  const float* wrow0 = W + (size_t)(nT * 64 + laneLo) * D_MODEL;

  v8f acc[4] = {v8f{}, v8f{}, v8f{}, v8f{}};

  for (int kb = 0; kb < D_MODEL; kb += 32) {
    // ---- A fragment (16x32 MxK): lane=M, two 8-half K-runs per half-wave
    v16h a;
    {
      const int k0 = kb + 8 * laneHi;
      const float4 x0 = *(const float4*)(xrow + k0);
      const float4 x1 = *(const float4*)(xrow + k0 + 4);
      const float4 x2 = *(const float4*)(xrow + k0 + 16);
      const float4 x3 = *(const float4*)(xrow + k0 + 20);
      a[0]=(_Float16)x0.x;  a[1]=(_Float16)x0.y;  a[2]=(_Float16)x0.z;  a[3]=(_Float16)x0.w;
      a[4]=(_Float16)x1.x;  a[5]=(_Float16)x1.y;  a[6]=(_Float16)x1.z;  a[7]=(_Float16)x1.w;
      a[8]=(_Float16)x2.x;  a[9]=(_Float16)x2.y;  a[10]=(_Float16)x2.z; a[11]=(_Float16)x2.w;
      a[12]=(_Float16)x3.x; a[13]=(_Float16)x3.y; a[14]=(_Float16)x3.z; a[15]=(_Float16)x3.w;
    }
    // ---- 4 N-subtiles share the A fragment
#pragma unroll
    for (int t = 0; t < 4; ++t) {
      const float* wrow = wrow0 + (size_t)t * 16 * D_MODEL;
      v16h bf;
      const int k1 = kb + 8 * laneHi;
      const float4 w0 = *(const float4*)(wrow + k1);
      const float4 w1 = *(const float4*)(wrow + k1 + 4);
      const float4 w2 = *(const float4*)(wrow + k1 + 16);
      const float4 w3 = *(const float4*)(wrow + k1 + 20);
      bf[0]=(_Float16)w0.x;  bf[1]=(_Float16)w0.y;  bf[2]=(_Float16)w0.z;  bf[3]=(_Float16)w0.w;
      bf[4]=(_Float16)w1.x;  bf[5]=(_Float16)w1.y;  bf[6]=(_Float16)w1.z;  bf[7]=(_Float16)w1.w;
      bf[8]=(_Float16)w2.x;  bf[9]=(_Float16)w2.y;  bf[10]=(_Float16)w2.z; bf[11]=(_Float16)w2.w;
      bf[12]=(_Float16)w3.x; bf[13]=(_Float16)w3.y; bf[14]=(_Float16)w3.z; bf[15]=(_Float16)w3.w;
      acc[t] = wmma_f16(a, bf, acc[t]);
    }
  }

  // D layout: vgpr r -> M = r + 8*laneHi, N = laneLo (per 16-wide subtile)
#pragma unroll
  for (int t = 0; t < 4; ++t) {
    const int n = nT * 64 + t * 16 + laneLo;
    const float bn = bias[n];
#pragma unroll
    for (int r = 0; r < 8; ++r) {
      const int mo = mT * 16 + r + 8 * laneHi;
      out[(size_t)mo * D_MODEL + n] = (_Float16)(acc[t][r] + bn);
    }
  }
}

// ---------------------------------------------------------------------------
// Kernel 2: flash-attention per (batch, head, 16-query tile). One wave/block.
// K and V 32x64 tiles are TDM double-buffered into LDS (s_wait_tensorcnt).
// ---------------------------------------------------------------------------
__global__ __launch_bounds__(32) void attn_kernel(
    const _Float16* __restrict__ qbuf, const _Float16* __restrict__ kbuf,
    const _Float16* __restrict__ vbuf, float* __restrict__ out)
{
  __shared__ float sS[16 * 32];                          // score tile (f32)
  __shared__ __align__(16) _Float16 sP[16 * 32];         // exp(P) tile (f16)
  __shared__ __align__(16) _Float16 sK[2][32 * 64];      // K tiles (ping-pong)
  __shared__ __align__(16) _Float16 sV[2][32 * 64];      // V tiles (ping-pong)
  __shared__ float row_max[16], row_sum[16], row_scale[16];

  const int lane   = threadIdx.x;
  const int laneLo = lane & 15;
  const int laneHi = lane >> 4;
  const int qt = blockIdx.x;     // 0..63
  const int h  = blockIdx.y;     // 0..11
  const int b  = blockIdx.z;     // 0..7
  const size_t rowBase = (size_t)b * SEQ;
  const size_t headOff = (size_t)h * D_K;

  if (lane < 16) { row_max[lane] = -__builtin_inff(); row_sum[lane] = 0.0f; }
  __syncthreads();

  // ---- Q A-fragments held in registers for the whole key loop
  const _Float16* qrow =
      qbuf + (rowBase + qt * 16 + laneLo) * D_MODEL + headOff;
  const v16h aq0 = hcat(*(const v8h*)(qrow +      8 * laneHi),
                        *(const v8h*)(qrow + 16 + 8 * laneHi));
  const v16h aq1 = hcat(*(const v8h*)(qrow + 32 + 8 * laneHi),
                        *(const v8h*)(qrow + 48 + 8 * laneHi));

  v8f acc[4] = {v8f{}, v8f{}, v8f{}, v8f{}};

  // ---- prologue: stage key-block 0
  stage_tile(kbuf + rowBase * D_MODEL + headOff, sK[0], lane);
  stage_tile(vbuf + rowBase * D_MODEL + headOff, sV[0], lane);

  int ib = 0;
  for (int kb = 0; kb < SEQ; kb += 32, ib ^= 1) {
    const bool have_next = (kb + 32) < SEQ;
    if (have_next) {   // issue async loads for the next block (other buffer)
      stage_tile(kbuf + (rowBase + kb + 32) * D_MODEL + headOff, sK[ib ^ 1], lane);
      stage_tile(vbuf + (rowBase + kb + 32) * D_MODEL + headOff, sV[ib ^ 1], lane);
      wait_tiles_pending_2();     // current pair (issued earlier) is complete
    } else {
      wait_tiles_pending_0();
    }
    __syncthreads();

    // ---- scores: two 16-key subtiles, each K=64 reduction = 2 WMMAs
#pragma unroll
    for (int j = 0; j < 2; ++j) {
      const _Float16* kr = sK[ib] + (j * 16 + laneLo) * 64;
      const v16h bk0 = hcat(*(const v8h*)(kr + 8 * laneHi),
                            *(const v8h*)(kr + 16 + 8 * laneHi));
      const v16h bk1 = hcat(*(const v8h*)(kr + 32 + 8 * laneHi),
                            *(const v8h*)(kr + 48 + 8 * laneHi));
      v8f sc = {};
      sc = wmma_f16(aq0, bk0, sc);
      sc = wmma_f16(aq1, bk1, sc);
#pragma unroll
      for (int r = 0; r < 8; ++r)
        sS[(r + 8 * laneHi) * 32 + j * 16 + laneLo] = sc[r] * 0.125f;
    }
    __syncthreads();

    // ---- online softmax: two lanes per row (laneHi picks 16-col half)
    {
      const int m = laneLo;
      float bm = -__builtin_inff();
#pragma unroll
      for (int e = 0; e < 16; ++e)
        bm = fmaxf(bm, sS[m * 32 + laneHi * 16 + e]);
      bm = fmaxf(bm, __shfl_xor(bm, 16));
      const float oldmax = row_max[m];
      const float newmax = fmaxf(oldmax, bm);
      const float factor = __expf(oldmax - newmax);
      float lsum = 0.0f;
#pragma unroll
      for (int e = 0; e < 16; ++e) {
        const float p = __expf(sS[m * 32 + laneHi * 16 + e] - newmax);
        lsum += p;
        sP[m * 32 + laneHi * 16 + e] = (_Float16)p;
      }
      lsum += __shfl_xor(lsum, 16);
      if (laneHi == 0) {
        row_sum[m]   = row_sum[m] * factor + lsum;
        row_max[m]   = newmax;
        row_scale[m] = factor;
      }
    }
    __syncthreads();

    // ---- rescale running context accumulators (row = r + 8*laneHi)
#pragma unroll
    for (int r = 0; r < 8; ++r) {
      const float f = row_scale[r + 8 * laneHi];
      acc[0][r] *= f; acc[1][r] *= f; acc[2][r] *= f; acc[3][r] *= f;
    }

    // ---- P A-fragment from LDS (16x32, K = key offset in block)
    const v16h ap = hcat(*(const v8h*)(sP + laneLo * 32 +      8 * laneHi),
                         *(const v8h*)(sP + laneLo * 32 + 16 + 8 * laneHi));

    // ---- P x V: 4 feature tiles of 16
#pragma unroll
    for (int ft = 0; ft < 4; ++ft) {
      v16h bv;
#pragma unroll
      for (int e = 0; e < 16; ++e)
        bv[e] = sV[ib][(8 * laneHi + e + (e & 8)) * 64 + ft * 16 + laneLo];
      acc[ft] = wmma_f16(ap, bv, acc[ft]);
    }
    __syncthreads();
  }

  // ---- epilogue: ctx / row_sum, written to [B,S,H*Dk] fp32
#pragma unroll
  for (int r = 0; r < 8; ++r) {
    const int mr = r + 8 * laneHi;
    const float inv = 1.0f / row_sum[mr];
    float* orow = out + (rowBase + qt * 16 + mr) * D_MODEL + headOff;
    orow[     laneLo] = acc[0][r] * inv;
    orow[16 + laneLo] = acc[1][r] * inv;
    orow[32 + laneLo] = acc[2][r] * inv;
    orow[48 + laneLo] = acc[3][r] * inv;
  }
}

extern "C" void kernel_launch(void* const* d_in, const int* in_sizes, int n_in,
                              void* d_out, int out_size, void* d_ws, size_t ws_size,
                              hipStream_t stream) {
  const float* Q  = (const float*)d_in[0];
  const float* Wq = (const float*)d_in[1];
  const float* bq = (const float*)d_in[2];
  const float* Wk = (const float*)d_in[3];
  const float* bk = (const float*)d_in[4];
  const float* Wv = (const float*)d_in[5];
  const float* bv = (const float*)d_in[6];
  float* out = (float*)d_out;

  const size_t elems = (size_t)M_TOTAL * D_MODEL;   // 6,291,456 f16 each
  _Float16* qbuf = (_Float16*)d_ws;
  _Float16* kbuf = qbuf + elems;
  _Float16* vbuf = kbuf + elems;

  dim3 gProj(D_MODEL / 64, M_TOTAL / 16, 3);        // 12 x 512 x 3 tiles
  qkv_proj_kernel<<<gProj, 32, 0, stream>>>(Q, Wq, bq, Wk, bk, Wv, bv,
                                            qbuf, kbuf, vbuf);

  dim3 gAttn(SEQ / 16, N_HEADS, BATCH);             // 64 x 12 x 8
  attn_kernel<<<gAttn, 32, 0, stream>>>(qbuf, kbuf, vbuf, out);
}
